// MultiHeadAttention_14061722927375
// MI455X (gfx1250) — matrix operand order repaired and checked by reading, not compiled, not run on hardware
//
#include <hip/hip_runtime.h>

// ---------------------------------------------------------------------------
// MI455X (gfx1250) multi-head attention, bf16 WMMA + flash softmax.
// All GEMM-side operand panels staged through LDS with CDNA5 async
// global->LDS copies, double buffered, raw split barriers (no vmem fence).
// ---------------------------------------------------------------------------

typedef __bf16 bf16;
typedef __attribute__((ext_vector_type(16))) __bf16 v16bf;
typedef __attribute__((ext_vector_type(8)))  __bf16 v8bf;
typedef __attribute__((ext_vector_type(8)))  float  v8f;

#define BB   4
#define SS   2048
#define DIMM 768
#define NH   16
#define DH   48     // head dim
#define DHP  64     // head dim padded to 2x K=32

#define BROW 40     // GEMM LDS row pitch (32 data + 8 pad halves) -> 80B
#define KROW 72     // attn K-panel row pitch (64 data + 8 pad halves) -> 144B
#define VROW 40     // attn V-panel row pitch (32 data + 8 pad halves) -> 80B
#define KSTEPS (DIMM / 32)   // 24

// D = A*B + C, bf16 inputs, f32 accumulate
__device__ __forceinline__ v8f wmma_bf16(v16bf a, v16bf b, v8f c) {
  return __builtin_amdgcn_wmma_f32_16x16x32_bf16(false, a, false, b, (short)0, c,
                                                 false, false);
}

__device__ __forceinline__ v16bf combine16(v8bf lo, v8bf hi) {
  v16bf r;
#pragma unroll
  for (int i = 0; i < 8; ++i) { r[i] = lo[i]; r[i + 8] = hi[i]; }
  return r;
}

// A operand (16 x 32 tile of a row-major MxK source, row stride ld).
// lanes 0-15:  M=lane,    K = {k0+0..7,  k0+16..23}
// lanes 16-31: M=lane-16, K = {k0+8..15, k0+24..31}
__device__ __forceinline__ v16bf ld_matA(const bf16* __restrict__ base, int ld,
                                         int k0, int lane) {
  const bf16* p = base + (size_t)(lane & 15) * ld + k0 + ((lane >> 4) << 3);
  return combine16(*(const v8bf*)p, *(const v8bf*)(p + 16));
}

// B operand from an LDS panel with row pitch `pitch` (halves).
// lanes 0-15: N=row0+lane, K={k0+0..7, k0+16..23} ; lanes 16-31: N=row0+lane-16, K={k0+8..15, k0+24..31}
__device__ __forceinline__ v16bf ld_matB_lds(const bf16* buf, int pitch,
                                             int row0, int k0, int ln, int hi) {
  const bf16* p = buf + (row0 + ln) * pitch + k0 + hi * 8;
  return combine16(*(const v8bf*)p, *(const v8bf*)(p + 16));
}

// ---------------------------------------------------------------------------
// CDNA5 async global -> LDS + raw workgroup barrier
// ---------------------------------------------------------------------------
__device__ __forceinline__ unsigned lds_off32(const void* p) {
  return (unsigned)(unsigned long long)(uintptr_t)p;  // low 32 bits = LDS offset
}

__device__ __forceinline__ void async_b128(const bf16* src, bf16* dst) {
  asm volatile("global_load_async_to_lds_b128 %0, %1, off"
               :: "v"(lds_off32(dst)), "v"((unsigned long long)(uintptr_t)src)
               : "memory");
}

__device__ __forceinline__ void wait_async0() {
  asm volatile("s_wait_asynccnt 0x0" ::: "memory");
}

// Plain split barrier: producers already drained their own ASYNCcnt, so no
// vmem-fence (s_wait_loadcnt 0) is needed -- keeps A prefetches in flight.
__device__ __forceinline__ void wg_barrier() {
  asm volatile("s_barrier_signal -1\n\ts_barrier_wait -1" ::: "memory");
}

// ---------------------------------------------------------------------------
// GEMM: stage one 64(N) x 32(K) bf16 weight panel (256 x b128 async copies)
// ---------------------------------------------------------------------------
__device__ __forceinline__ void stage_W(const bf16* __restrict__ Wn0, int k0,
                                        bf16* __restrict__ buf, int tid) {
  int row = tid >> 2, chunk = tid & 3;   // 64 rows x 4 chunks of 8 halves
  async_b128(Wn0 + (size_t)row * DIMM + k0 + chunk * 8,
             buf + row * BROW + chunk * 8);
}

// Sequential load->WMMA form: one B temp live at a time (minimizes VGPR
// live-range interference with the accumulators, avoids allocator copies).
__device__ __forceinline__ void wmma4(v8f* acc, v16bf a, const bf16* buf,
                                      int ln, int hi) {
#pragma unroll
  for (int nt = 0; nt < 4; ++nt) {
    v16bf b = ld_matB_lds(buf, BROW, nt * 16, 0, ln, hi);
    acc[nt] = wmma_bf16(a, b, acc[nt]);
  }
}

// acc[4] += Arows(16 x 768) @ W(n0..n0+63, :)^T ; explicit ping-pong, no
// register rotation, copy of panel k+1 overlapped with WMMAs on panel k.
__device__ __forceinline__ void gemm_mainloop(const bf16* __restrict__ Arows,
                                              const bf16* __restrict__ Wn0,
                                              bf16* __restrict__ lds2,
                                              int lane, int tid, v8f acc[4]) {
  int hi = lane >> 4, ln = lane & 15;
  bf16* buf0 = lds2;
  bf16* buf1 = lds2 + 64 * BROW;
  stage_W(Wn0, 0, buf0, tid);
  v16bf a0 = ld_matA(Arows, DIMM, 0, lane);
  v16bf a1;
  for (int ks = 0; ks < KSTEPS - 2; ks += 2) {
    wait_async0(); wg_barrier();
    stage_W(Wn0, (ks + 1) * 32, buf1, tid);
    a1 = ld_matA(Arows, DIMM, (ks + 1) * 32, lane);
    wmma4(acc, a0, buf0, ln, hi);
    wait_async0(); wg_barrier();
    stage_W(Wn0, (ks + 2) * 32, buf0, tid);
    a0 = ld_matA(Arows, DIMM, (ks + 2) * 32, lane);
    wmma4(acc, a1, buf1, ln, hi);
  }
  wait_async0(); wg_barrier();
  stage_W(Wn0, (KSTEPS - 1) * 32, buf1, tid);
  a1 = ld_matA(Arows, DIMM, (KSTEPS - 1) * 32, lane);
  wmma4(acc, a0, buf0, ln, hi);
  wait_async0(); wg_barrier();
  wmma4(acc, a1, buf1, ln, hi);
}

// ---------------------------------------------------------------------------
// Kernel 1: fp32 -> bf16 conversion of x and the four weight matrices.
// ---------------------------------------------------------------------------
__global__ void convert_kernel(const float* __restrict__ x,
                               const float* __restrict__ Wq,
                               const float* __restrict__ Wk,
                               const float* __restrict__ Wv,
                               const float* __restrict__ Wo,
                               bf16* __restrict__ xb, bf16* __restrict__ wb) {
  const int NX = BB * SS * DIMM;
  const int NW = DIMM * DIMM;
  const int total = NX + 4 * NW;
  for (int i = blockIdx.x * blockDim.x + threadIdx.x; i < total;
       i += gridDim.x * blockDim.x) {
    if (i < NX) {
      xb[i] = (bf16)x[i];
    } else {
      int j = i - NX;
      int w = j / NW, o = j % NW;
      const float* src = (w == 0) ? Wq : (w == 1) ? Wk : (w == 2) ? Wv : Wo;
      wb[(size_t)w * NW + o] = (bf16)src[o];
    }
  }
}

__global__ void zero_kernel(uint4* __restrict__ p, int n16) {
  uint4 z = make_uint4(0u, 0u, 0u, 0u);
  for (int i = blockIdx.x * blockDim.x + threadIdx.x; i < n16;
       i += gridDim.x * blockDim.x)
    p[i] = z;
}

// ---------------------------------------------------------------------------
// Kernel 2: fused QKV projection. Block = 128(M) x 64(N) tile of one proj.
//   Qp/Kp: [B,H,S,64] bf16 (dh padded, pads pre-zeroed)
//   Vt   : [B,H,48,S] bf16 (transposed for the P@V B-operand)
// Grid: 3 proj * 64 Mblocks * 12 Ngroups = 2304 blocks of 256.
// ---------------------------------------------------------------------------
__global__ void qkv_kernel(const bf16* __restrict__ xb,
                           const bf16* __restrict__ wb,
                           const float* __restrict__ bq,
                           const float* __restrict__ bk,
                           const float* __restrict__ bv,
                           bf16* __restrict__ Qp, bf16* __restrict__ Kp,
                           bf16* __restrict__ Vt) {
  __shared__ __align__(16) bf16 Bst[2 * 64 * BROW];   // 10 KB
  const int MB = (BB * SS) / 128;  // 64
  const int NG = DIMM / 64;        // 12
  int bid = blockIdx.x;
  int proj = bid / (MB * NG);
  int rem  = bid % (MB * NG);
  int mb = rem / NG, ng = rem % NG;
  int tid = threadIdx.x;
  int wslot = tid >> 5, lane = tid & 31;
  int m0 = mb * 128 + wslot * 16;
  int n0 = ng * 64;
  const bf16* Wm = wb + (size_t)proj * DIMM * DIMM;

  v8f acc[4] = {};
  gemm_mainloop(xb + (size_t)m0 * DIMM, Wm + (size_t)n0 * DIMM, Bst,
                lane, tid, acc);

  const float* bias = (proj == 0) ? bq : (proj == 1) ? bk : bv;
  int hi = lane >> 4, ln = lane & 15;
#pragma unroll
  for (int nt = 0; nt < 4; ++nt) {
    int col = n0 + nt * 16 + ln;          // 16-col tile stays within one head
    float bias_c = bias[col];
    int h = col / DH, dh = col % DH;
#pragma unroll
    for (int r = 0; r < 8; ++r) {
      int m = m0 + r + 8 * hi;
      int b = m >> 11, s = m & (SS - 1);
      float v = acc[nt][r] + bias_c;
      if (proj == 0)
        Qp[(((size_t)(b * NH + h)) * SS + s) * DHP + dh] = (bf16)v;
      else if (proj == 1)
        Kp[(((size_t)(b * NH + h)) * SS + s) * DHP + dh] = (bf16)v;
      else
        Vt[(((size_t)(b * NH + h)) * DH + dh) * SS + s] = (bf16)v;
    }
  }
}

// ---------------------------------------------------------------------------
// Kernel 3: flash attention. Block = 8 waves, all on the SAME (b,h); each
// wave owns a 16-query tile. K/V chunks (32 keys) are async-staged into LDS
// once per block (8x L2 traffic cut), double buffered; per-wave online
// softmax; P goes through a per-wave LDS bounce (C-layout -> A-layout).
// Grid: 64 bh * 16 tile-groups = 1024 blocks of 256.
// ---------------------------------------------------------------------------
__global__ void attn_kernel(const bf16* __restrict__ Qp,
                            const bf16* __restrict__ Kp,
                            const bf16* __restrict__ Vt,
                            bf16* __restrict__ aout) {
  __shared__ __align__(16) bf16 Kst[2 * 32 * KROW];   // 9.2 KB
  __shared__ __align__(16) bf16 Vst[2 * 48 * VROW];   // 7.7 KB
  __shared__ __align__(16) bf16 plds[8][16 * 32];     // 8 KB
  int tid = threadIdx.x;
  int wslot = tid >> 5, lane = tid & 31;
  int bh  = blockIdx.x >> 4;            // 0..63
  int grp = blockIdx.x & 15;
  int m0  = (grp * 8 + wslot) * 16;
  int b = bh / NH, h = bh % NH;

  const bf16* Qb = Qp + ((size_t)bh * SS + m0) * DHP;
  const bf16* Kb = Kp + (size_t)bh * SS * DHP;
  const bf16* Vb = Vt + (size_t)bh * DH * SS;
  int hi = lane >> 4, ln = lane & 15;

  // chunk staging: K panel 32 rows x 64 halves (256 x b128),
  //                V panel 48 rows x 32 halves (192 x b128)
  int krow = tid >> 3, kc = tid & 7;
  int vrow = tid >> 2, vc = tid & 3;

  async_b128(Kb + (size_t)krow * DHP + kc * 8, Kst + krow * KROW + kc * 8);
  if (tid < 192)
    async_b128(Vb + (size_t)vrow * SS + vc * 8, Vst + vrow * VROW + vc * 8);

  v16bf aQ0 = ld_matA(Qb, DHP, 0, lane);
  v16bf aQ1 = ld_matA(Qb, DHP, 32, lane);  // cols 48..63 are zero pads

  v8f o0 = {}, o1 = {}, o2 = {};
  float mrow[8], lrow[8];
#pragma unroll
  for (int r = 0; r < 8; ++r) { mrow[r] = -1e30f; lrow[r] = 0.f; }
  bf16* P = plds[wslot];
  const float scale = 0.14433756729740643f;  // 1/sqrt(48)

  for (int s0 = 0; s0 < SS; s0 += 32) {
    int cur = (s0 >> 5) & 1;
    bf16* Kc = Kst + cur * (32 * KROW);
    bf16* Vc = Vst + cur * (48 * VROW);
    wait_async0(); wg_barrier();
    if (s0 + 32 < SS) {
      bf16* Kn = Kst + (cur ^ 1) * (32 * KROW);
      bf16* Vn = Vst + (cur ^ 1) * (48 * VROW);
      async_b128(Kb + (size_t)(s0 + 32 + krow) * DHP + kc * 8,
                 Kn + krow * KROW + kc * 8);
      if (tid < 192)
        async_b128(Vb + (size_t)vrow * SS + (s0 + 32) + vc * 8,
                   Vn + vrow * VROW + vc * 8);
    }

    // scores: two 16-key tiles, K-dim = padded head dim 64 (sequential
    // load->WMMA to keep one B temp live at a time)
    v8f c0 = {}, c1 = {};
    {
      v16bf kb = ld_matB_lds(Kc, KROW, 0, 0, ln, hi);
      c0 = wmma_bf16(aQ0, kb, c0);
      kb = ld_matB_lds(Kc, KROW, 0, 32, ln, hi);
      c0 = wmma_bf16(aQ1, kb, c0);
      kb = ld_matB_lds(Kc, KROW, 16, 0, ln, hi);
      c1 = wmma_bf16(aQ0, kb, c1);
      kb = ld_matB_lds(Kc, KROW, 16, 32, ln, hi);
      c1 = wmma_bf16(aQ1, kb, c1);
    }

    // online softmax; D-tile row r lives in one 16-lane half, so xor-shuffles
    // {1,2,4,8} reduce exactly over the key (N) dimension.
    float p0[8], p1[8];
#pragma unroll
    for (int r = 0; r < 8; ++r) {
      float x0 = c0[r] * scale, x1 = c1[r] * scale;
      float t = fmaxf(x0, x1);
      t = fmaxf(t, __shfl_xor(t, 1));
      t = fmaxf(t, __shfl_xor(t, 2));
      t = fmaxf(t, __shfl_xor(t, 4));
      t = fmaxf(t, __shfl_xor(t, 8));
      float mn = fmaxf(mrow[r], t);
      float al = __expf(mrow[r] - mn);
      float e0 = __expf(x0 - mn), e1 = __expf(x1 - mn);
      float ts = e0 + e1;
      ts += __shfl_xor(ts, 1);
      ts += __shfl_xor(ts, 2);
      ts += __shfl_xor(ts, 4);
      ts += __shfl_xor(ts, 8);
      lrow[r] = lrow[r] * al + ts;
      mrow[r] = mn;
      o0[r] *= al; o1[r] *= al; o2[r] *= al;
      p0[r] = e0;  p1[r] = e1;
    }

    // C-layout -> per-wave LDS row-major 16x32, reload in A-operand layout
    // (DS ops are in-order within a wave)
#pragma unroll
    for (int r = 0; r < 8; ++r) {
      int row = r + 8 * hi;
      P[row * 32 + ln]      = (bf16)p0[r];
      P[row * 32 + 16 + ln] = (bf16)p1[r];
    }
    v16bf aP;
    {
      const bf16* pr = P + ln * 32 + (hi << 3);
      aP = combine16(*(const v8bf*)pr, *(const v8bf*)(pr + 16));
    }

    // o += P @ V  (K = 32 keys, N = 48 dh as 3 tiles, from staged V panel)
    {
      v16bf vb = ld_matB_lds(Vc, VROW, 0, 0, ln, hi);
      o0 = wmma_bf16(aP, vb, o0);
      vb = ld_matB_lds(Vc, VROW, 16, 0, ln, hi);
      o1 = wmma_bf16(aP, vb, o1);
      vb = ld_matB_lds(Vc, VROW, 32, 0, ln, hi);
      o2 = wmma_bf16(aP, vb, o2);
    }
  }

#pragma unroll
  for (int r = 0; r < 8; ++r) {
    float inv = 1.f / lrow[r];
    int m = m0 + r + 8 * hi;
    size_t rowoff = ((size_t)b * SS + m) * DIMM + h * DH;
    aout[rowoff + ln]      = (bf16)(o0[r] * inv);
    aout[rowoff + 16 + ln] = (bf16)(o1[r] * inv);
    aout[rowoff + 32 + ln] = (bf16)(o2[r] * inv);
  }
}

// ---------------------------------------------------------------------------
// Kernel 4: output projection, f32 result: out = attn @ Wo.T + bo
// Grid: 64 Mblocks * 12 Ngroups = 768 blocks of 256.
// ---------------------------------------------------------------------------
__global__ void out_proj_kernel(const bf16* __restrict__ ain,
                                const bf16* __restrict__ Wo,
                                const float* __restrict__ bo,
                                float* __restrict__ out) {
  __shared__ __align__(16) bf16 Bst[2 * 64 * BROW];
  const int NG = DIMM / 64;
  int mb = blockIdx.x / NG, ng = blockIdx.x % NG;
  int tid = threadIdx.x;
  int wslot = tid >> 5, lane = tid & 31;
  int m0 = mb * 128 + wslot * 16;
  int n0 = ng * 64;

  v8f acc[4] = {};
  gemm_mainloop(ain + (size_t)m0 * DIMM, Wo + (size_t)n0 * DIMM, Bst,
                lane, tid, acc);

  int hi = lane >> 4, ln = lane & 15;
#pragma unroll
  for (int nt = 0; nt < 4; ++nt) {
    int col = n0 + nt * 16 + ln;
    float bias_c = bo[col];
#pragma unroll
    for (int r = 0; r < 8; ++r) {
      int m = m0 + r + 8 * hi;
      out[(size_t)m * DIMM + col] = acc[nt][r] + bias_c;
    }
  }
}

// ---------------------------------------------------------------------------
extern "C" void kernel_launch(void* const* d_in, const int* in_sizes, int n_in,
                              void* d_out, int out_size, void* d_ws,
                              size_t ws_size, hipStream_t stream) {
  const float* x  = (const float*)d_in[0];
  const float* Wq = (const float*)d_in[1];
  const float* bq = (const float*)d_in[2];
  const float* Wk = (const float*)d_in[3];
  const float* bk = (const float*)d_in[4];
  const float* Wv = (const float*)d_in[5];
  const float* bv = (const float*)d_in[6];
  const float* Wo = (const float*)d_in[7];
  const float* bo = (const float*)d_in[8];
  float* out = (float*)d_out;

  // workspace layout (bf16 elements); ~63.4 MB total
  bf16* xb = (bf16*)d_ws;                                 // B*S*DIM
  bf16* wb = xb + (size_t)BB * SS * DIMM;                 // 4 * DIM*DIM
  bf16* Qp = wb + (size_t)4 * DIMM * DIMM;                // B*H*S*DHP
  bf16* Kp = Qp + (size_t)BB * NH * SS * DHP;             // B*H*S*DHP
  bf16* Vt = Kp + (size_t)BB * NH * SS * DHP;             // B*H*DH*S
  bf16* aout = xb;  // alias: xb is dead after the QKV GEMM

  convert_kernel<<<dim3(4096), dim3(256), 0, stream>>>(x, Wq, Wk, Wv, Wo, xb, wb);

  int n16 = (int)(((size_t)2 * BB * NH * SS * DHP * sizeof(bf16)) / 16);
  zero_kernel<<<dim3(2048), dim3(256), 0, stream>>>((uint4*)Qp, n16);

  // 3 proj * 64 Mblocks * 12 Ngroups
  qkv_kernel<<<dim3(2304), dim3(256), 0, stream>>>(xb, wb, bq, bk, bv, Qp, Kp, Vt);

  // 64 bh * 16 tile-groups
  attn_kernel<<<dim3(1024), dim3(256), 0, stream>>>(Qp, Kp, Vt, aout);

  // 64 Mblocks * 12 Ngroups
  out_proj_kernel<<<dim3(768), dim3(256), 0, stream>>>(
      aout, wb + (size_t)3 * DIMM * DIMM, bo, out);
}
